// RWKVCM_22866405884264
// MI455X (gfx1250) — matrix operand order repaired and checked by reading, not compiled, not run on hardware
//
#include <hip/hip_runtime.h>
#include <hip/hip_bf16.h>
#include <stdint.h>

#define B_    2
#define T_    1024
#define D_    1024
#define V_    50257
#define A_    256
#define VPAD_ 50432        // 394 * 128, N padded to tile multiple
#define M_    (B_ * T_)    // 2048 flattened rows

#define KSTEP 32
#define PITCH 40           // LDS tile row pitch in bf16 elements (80B, 16B-aligned, bank-spread)

typedef __attribute__((ext_vector_type(16))) __bf16 v16bf;
typedef __attribute__((ext_vector_type(8)))  __bf16 v8bf;
typedef __attribute__((ext_vector_type(8)))  float  v8f;

union AF16 { v16bf v; v8bf h[2]; };

#if __has_builtin(__builtin_amdgcn_global_load_async_to_lds_b128)
#define HAVE_ASYNC_LDS 1
#endif

// copy 16B global -> LDS; async (ASYNCcnt-tracked) when available.
// `l` is a generic pointer that provably points into LDS (__shared__).
__device__ __forceinline__ void cp16(const __bf16* g, unsigned short* l) {
#ifdef HAVE_ASYNC_LDS
  typedef int vsi4 __attribute__((vector_size(4 * sizeof(int))));
  __builtin_amdgcn_global_load_async_to_lds_b128((vsi4*)g, (vsi4*)l, 0, 0);
#else
  *(v8bf*)l = *(const v8bf*)g;
#endif
}

__device__ __forceinline__ void wait_async0() {
#if __has_builtin(__builtin_amdgcn_s_wait_asynccnt)
  __builtin_amdgcn_s_wait_asynccnt(0);
#else
  asm volatile("s_wait_asynccnt 0x0" ::: "memory");
#endif
}

// round-to-nearest-even f32 -> bf16 (inputs are finite gaussians; no NaN path)
__device__ __forceinline__ unsigned short f2bf(float f) {
  unsigned int u = __float_as_uint(f);
  u += 0x7fffu + ((u >> 16) & 1u);
  return (unsigned short)(u >> 16);
}

// A fragment, 16x32 bf16 (MxK), source row-major with K contiguous.
// lane l: row = l&15, chunks of 8 K at (l>>4)*8 and (l>>4)*8+16.
__device__ __forceinline__ v16bf ld_afrag(const __bf16* base, int pitch, int lane) {
  const int row = lane & 15;
  const int kg  = (lane >> 4) << 3;
  const __bf16* p = base + (size_t)row * pitch + kg;
  AF16 f;
  f.h[0] = *(const v8bf*)(p);
  f.h[1] = *(const v8bf*)(p + 16);
  return f.v;
}

// B fragment, 32x16 bf16 (KxN), where B = W^T and W is row-major [N x K]:
// lane l: column = l&15 (a W row, contiguous in K), chunks of 8 K at (l>>4)*8 and (l>>4)*8+16.
__device__ __forceinline__ v16bf ld_bfrag_wt(const __bf16* base, int pitch, int lane) {
  const int col = lane & 15;
  const int kb  = (lane >> 4) << 3;
  const __bf16* p = base + (size_t)col * pitch + kb;
  AF16 f;
  f.h[0] = *(const v8bf*)(p);
  f.h[1] = *(const v8bf*)(p + 16);
  return f.v;
}

// ---------------- conversion kernels ----------------

__global__ void cvt_f32_bf16(const float* __restrict__ in,
                             unsigned short* __restrict__ out, int n4) {
  int i = blockIdx.x * blockDim.x + threadIdx.x;
  if (i < n4) {
    float4 f = ((const float4*)in)[i];
    ushort4 o;
    o.x = f2bf(f.x); o.y = f2bf(f.y); o.z = f2bf(f.z); o.w = f2bf(f.w);
    ((ushort4*)out)[i] = o;
  }
}

// W_head -> bf16 with zero padding of rows [V_, VPAD_)
__global__ void cvt_whead(const float* __restrict__ W,
                          unsigned short* __restrict__ out) {
  int i = blockIdx.x * blockDim.x + threadIdx.x;   // over VPAD_*D_/4
  const int n4 = VPAD_ * D_ / 4;
  if (i < n4) {
    int row = i >> 8;                              // (i*4)/D_, D_=1024
    ushort4 o;
    if (row < V_) {
      float4 f = ((const float4*)W)[i];
      o.x = f2bf(f.x); o.y = f2bf(f.y); o.z = f2bf(f.z); o.w = f2bf(f.w);
    } else {
      o.x = 0; o.y = 0; o.z = 0; o.w = 0;
    }
    ((ushort4*)out)[i] = o;
  }
}

// ---------------- q/k projection GEMM (direct-global WMMA) ----------------
__global__ __launch_bounds__(256) void qk_gemm(const __bf16* __restrict__ xb,
                                               const __bf16* __restrict__ wqb,
                                               const __bf16* __restrict__ wkb,
                                               unsigned short* __restrict__ qb,
                                               unsigned short* __restrict__ kb) {
  const __bf16* W = blockIdx.z ? wkb : wqb;
  unsigned short* O = blockIdx.z ? kb : qb;
  const int lane = threadIdx.x & 31;
  const int wave = threadIdx.x >> 5;
  const int tileM = blockIdx.y * 128 + (wave >> 2) * 64;
  const int tileN = blockIdx.x * 128 + (wave & 3) * 32;

  const __bf16* Ab = xb + (size_t)tileM * D_;
  const __bf16* Bb = W  + (size_t)tileN * D_;

  v8f acc[4][2] = {};
  #pragma unroll 2
  for (int k = 0; k < D_; k += KSTEP) {
    v16bf a[4], b[2];
    #pragma unroll
    for (int i = 0; i < 4; ++i) a[i] = ld_afrag(Ab + (size_t)i * 16 * D_ + k, D_, lane);
    #pragma unroll
    for (int j = 0; j < 2; ++j) b[j] = ld_bfrag_wt(Bb + (size_t)j * 16 * D_ + k, D_, lane);
    #pragma unroll
    for (int i = 0; i < 4; ++i)
      #pragma unroll
      for (int j = 0; j < 2; ++j)
        acc[i][j] = __builtin_amdgcn_wmma_f32_16x16x32_bf16(
            false, a[i], false, b[j], (short)0, acc[i][j], false, false);
  }
  const int ro = (lane >> 4) * 8;
  const int co = lane & 15;
  #pragma unroll
  for (int i = 0; i < 4; ++i)
    #pragma unroll
    for (int j = 0; j < 2; ++j) {
      const int n = tileN + j * 16 + co;
      #pragma unroll
      for (int r = 0; r < 8; ++r) {
        const size_t m = (size_t)(tileM + i * 16 + ro + r);
        O[m * A_ + n] = f2bf(acc[i][j][r]);
      }
    }
}

// ---------------- copy-attention score GEMM ----------------
// per batch: c[t][s] = (q[t,:] . k[s,:]) / A_, masked to 0 for s > t
__global__ __launch_bounds__(256) void c_gemm(const __bf16* __restrict__ qb,
                                              const __bf16* __restrict__ kb,
                                              float* __restrict__ c) {
  const int bz = blockIdx.z;
  const int lane = threadIdx.x & 31;
  const int wave = threadIdx.x >> 5;
  const int tileT = blockIdx.y * 128 + (wave >> 2) * 64;
  const int tileS = blockIdx.x * 128 + (wave & 3) * 32;

  const __bf16* Ab = qb + (size_t)bz * T_ * A_ + (size_t)tileT * A_;
  const __bf16* Bb = kb + (size_t)bz * T_ * A_ + (size_t)tileS * A_;
  float* cb = c + (size_t)bz * T_ * T_;

  v8f acc[4][2] = {};
  #pragma unroll
  for (int k = 0; k < A_; k += KSTEP) {
    v16bf a[4], b[2];
    #pragma unroll
    for (int i = 0; i < 4; ++i) a[i] = ld_afrag(Ab + (size_t)i * 16 * A_ + k, A_, lane);
    #pragma unroll
    for (int j = 0; j < 2; ++j) b[j] = ld_bfrag_wt(Bb + (size_t)j * 16 * A_ + k, A_, lane);
    #pragma unroll
    for (int i = 0; i < 4; ++i)
      #pragma unroll
      for (int j = 0; j < 2; ++j)
        acc[i][j] = __builtin_amdgcn_wmma_f32_16x16x32_bf16(
            false, a[i], false, b[j], (short)0, acc[i][j], false, false);
  }
  const int ro = (lane >> 4) * 8;
  const int co = lane & 15;
  const float scale = 1.0f / (float)A_;
  #pragma unroll
  for (int i = 0; i < 4; ++i)
    #pragma unroll
    for (int j = 0; j < 2; ++j) {
      const int s = tileS + j * 16 + co;
      #pragma unroll
      for (int r = 0; r < 8; ++r) {
        const int t = tileT + i * 16 + ro + r;
        cb[(size_t)t * T_ + s] = (s <= t) ? acc[i][j][r] * scale : 0.0f;
      }
    }
}

// ---------------- main head GEMM: async-LDS staged, double buffered ----------------
// logits[m][v] = x[m,:] . W_head[v,:]  (bf16 inputs, f32 accumulate)
__global__ __launch_bounds__(256) void head_gemm(const __bf16* __restrict__ xb,
                                                 const __bf16* __restrict__ whb,
                                                 float* __restrict__ out) {
  __shared__ unsigned short smA[2][128 * PITCH];   // 128 x 32 bf16 tile, padded rows
  __shared__ unsigned short smB[2][128 * PITCH];

  const int tid  = threadIdx.x;
  const int lane = tid & 31;
  const int wave = tid >> 5;
  const int wm = (wave >> 2) * 64;                 // 2x4 wave grid over 128x128 tile
  const int wn = (wave & 3) * 32;
  const int tileM = blockIdx.y * 128;
  const int tileN = blockIdx.x * 128;

  const __bf16* Ag = xb  + (size_t)tileM * D_;
  const __bf16* Bg = whb + (size_t)tileN * D_;

  // stage one 128x32 A tile and one 128x32 B tile into LDS buffer `buf`.
  // 512 16B chunks per tile; 256 threads handle 2 chunks each, 4-lane-contiguous 64B runs.
  auto stage = [&](int buf, int k) {
    #pragma unroll
    for (int u = 0; u < 2; ++u) {
      const int c   = tid + u * 256;
      const int row = c >> 2;
      const int ch  = (c & 3) * 8;                 // element offset of 16B chunk
      cp16(Ag + (size_t)row * D_ + k + ch, &smA[buf][row * PITCH + ch]);
      cp16(Bg + (size_t)row * D_ + k + ch, &smB[buf][row * PITCH + ch]);
    }
  };

  stage(0, 0);
  wait_async0();
  __syncthreads();

  v8f acc[4][2] = {};
  #pragma unroll 2
  for (int k = 0; k < D_; k += KSTEP) {
    const int buf = (k >> 5) & 1;
    if (k + KSTEP < D_) stage(buf ^ 1, k + KSTEP);

    v16bf a[4], b[2];
    #pragma unroll
    for (int i = 0; i < 4; ++i)
      a[i] = ld_afrag((const __bf16*)&smA[buf][(wm + i * 16) * PITCH], PITCH, lane);
    #pragma unroll
    for (int j = 0; j < 2; ++j)
      b[j] = ld_bfrag_wt((const __bf16*)&smB[buf][(wn + j * 16) * PITCH], PITCH, lane);
    #pragma unroll
    for (int i = 0; i < 4; ++i)
      #pragma unroll
      for (int j = 0; j < 2; ++j)
        acc[i][j] = __builtin_amdgcn_wmma_f32_16x16x32_bf16(
            false, a[i], false, b[j], (short)0, acc[i][j], false, false);

    wait_async0();       // my async copies for buf^1 are done
    __syncthreads();     // everyone's copies visible; all reads of buf finished
  }

  const int ro = (lane >> 4) * 8;
  const int co = lane & 15;
  #pragma unroll
  for (int i = 0; i < 4; ++i)
    #pragma unroll
    for (int j = 0; j < 2; ++j) {
      const int n = tileN + wn + j * 16 + co;
      if (n < V_) {
        #pragma unroll
        for (int r = 0; r < 8; ++r) {
          const size_t m = (size_t)(tileM + wm + i * 16 + ro + r);
          out[m * (size_t)V_ + n] = acc[i][j][r];
        }
      }
    }
}

// ---------------- scatter of copy term ----------------
// logits[b, t, idx[b, s]] += c[b, t, s]   for s <= t (c is 0 above diagonal)
__global__ void scatter_copy(const float* __restrict__ c,
                             const int* __restrict__ idx,
                             float* __restrict__ out) {
  const int b = blockIdx.y;
  const int t = blockIdx.x;
  const float* row = c + ((size_t)b * T_ + t) * T_;
  const int* ib = idx + b * T_;
  float* orow = out + ((size_t)b * T_ + t) * (size_t)V_;
  for (int s = threadIdx.x; s <= t; s += blockDim.x) {
    atomicAdd(&orow[ib[s]], row[s]);
  }
}

extern "C" void kernel_launch(void* const* d_in, const int* in_sizes, int n_in,
                              void* d_out, int out_size, void* d_ws, size_t ws_size,
                              hipStream_t stream) {
  (void)in_sizes; (void)n_in; (void)out_size; (void)ws_size;
  const float* x      = (const float*)d_in[0];
  const int*   idx    = (const int*)d_in[1];
  const float* W_head = (const float*)d_in[2];
  const float* W_q    = (const float*)d_in[3];
  const float* W_k    = (const float*)d_in[4];
  float* out = (float*)d_out;

  char* ws = (char*)d_ws;
  size_t o = 0;
  __bf16* xb  = (__bf16*)(ws + o); o += (size_t)M_ * D_ * 2;
  __bf16* whb = (__bf16*)(ws + o); o += (size_t)VPAD_ * D_ * 2;
  __bf16* wqb = (__bf16*)(ws + o); o += (size_t)A_ * D_ * 2;
  __bf16* wkb = (__bf16*)(ws + o); o += (size_t)A_ * D_ * 2;
  __bf16* qb  = (__bf16*)(ws + o); o += (size_t)M_ * A_ * 2;
  __bf16* kb  = (__bf16*)(ws + o); o += (size_t)M_ * A_ * 2;
  float*  c   = (float*)(ws + o);  o += (size_t)B_ * T_ * T_ * 4;

  int n4;
  n4 = M_ * D_ / 4;
  cvt_f32_bf16<<<(n4 + 255) / 256, 256, 0, stream>>>(x, (unsigned short*)xb, n4);
  n4 = A_ * D_ / 4;
  cvt_f32_bf16<<<(n4 + 255) / 256, 256, 0, stream>>>(W_q, (unsigned short*)wqb, n4);
  cvt_f32_bf16<<<(n4 + 255) / 256, 256, 0, stream>>>(W_k, (unsigned short*)wkb, n4);
  n4 = VPAD_ * D_ / 4;
  cvt_whead<<<(n4 + 255) / 256, 256, 0, stream>>>(W_head, (unsigned short*)whb);

  qk_gemm<<<dim3(A_ / 128, M_ / 128, 2), 256, 0, stream>>>(
      xb, wqb, wkb, (unsigned short*)qb, (unsigned short*)kb);
  c_gemm<<<dim3(T_ / 128, T_ / 128, B_), 256, 0, stream>>>(qb, kb, c);
  head_gemm<<<dim3(VPAD_ / 128, M_ / 128), 256, 0, stream>>>(xb, whb, out);
  scatter_copy<<<dim3(T_, B_), 256, 0, stream>>>(c, idx, out);
}
